// LinearCrossAttention_44169443672900
// MI455X (gfx1250) — compile-verified, matrix operand order checked by reading, not run on hardware
//
#include <hip/hip_runtime.h>
#include <math.h>

typedef __attribute__((ext_vector_type(16))) _Float16 v16h;
typedef __attribute__((ext_vector_type(8)))  _Float16 v8h;
typedef __attribute__((ext_vector_type(8)))  float    v8f;
typedef __attribute__((ext_vector_type(4)))  unsigned int u32x4;
typedef __attribute__((ext_vector_type(4)))  int      i32x4;
typedef __attribute__((ext_vector_type(8)))  int      i32x8;
typedef _Float16 half_t;

#define EPS_ATTN 1e-6f
#define EPS_GN   1e-5f

// Tensor Data Mover availability / arity detection
#if defined(__has_builtin)
#if __has_builtin(__builtin_amdgcn_tensor_load_to_lds)
#define USE_TDM 1
#endif
#endif
#ifndef USE_TDM
#define USE_TDM 0
#endif
#if defined(__has_include)
#if __has_include(<hip/amd_detail/amd_gfx1250_TDM.h>)
#define TDM_6ARG 1
#else
#define TDM_6ARG 0
#endif
#else
#define TDM_6ARG 0
#endif

// Problem dims (fixed by reference)
static constexpr int  BATCH = 8;
static constexpr int  CCH   = 256;          // GEO_CH == SEM_CH
static constexpr int  NPIX  = 64 * 64;      // 4096
static constexpr long long CN  = (long long)CCH * NPIX;   // 1,048,576
static constexpr long long BCN = (long long)BATCH * CN;   // 8,388,608

// Padded LDS tile: 128 rows x 32 halfs, +8 halfs pad => 80B row stride
static constexpr int LDS_STRIDE = 40;   // halfs per row

// ---------------------------------------------------------------- utilities

__device__ __forceinline__ float block_reduce_sum(float v, float* sbuf) {
  __syncthreads();
  int t = threadIdx.x;
  sbuf[t] = v;
  __syncthreads();
  for (int s = blockDim.x >> 1; s > 0; s >>= 1) {
    if (t < s) sbuf[t] += sbuf[t + s];
    __syncthreads();
  }
  return sbuf[0];
}

__global__ void __launch_bounds__(256)
k_f32_to_f16(const float* __restrict__ in, half_t* __restrict__ out, int n) {
  int i = blockIdx.x * 256 + threadIdx.x;
  if (i < n) out[i] = (half_t)in[i];
}

// f_geo [B,C,4096] f32 -> XgeoT [B,4096,C] f16 (coalesced writes)
__global__ void __launch_bounds__(256)
k_transpose_to_f16(const float* __restrict__ in, half_t* __restrict__ out) {
  long long idx = (long long)blockIdx.x * 256 + threadIdx.x;  // [b][n][c]
  int c = (int)(idx & 255);
  int n = (int)((idx >> 8) & 4095);
  int b = (int)(idx >> 20);
  out[idx] = (half_t)in[(((long long)b * CCH + c) << 12) | n];
}

// Bilinear 2x upsample (half-pixel centers): f_sem[B,C,32,32] -> XsemT [B,4096,C] f16
__global__ void __launch_bounds__(256)
k_upsample_t(const float* __restrict__ src, half_t* __restrict__ dst) {
  long long idx = (long long)blockIdx.x * 256 + threadIdx.x;  // [b][n][c]
  int c = (int)(idx & 255);
  int n = (int)((idx >> 8) & 4095);
  int b = (int)(idx >> 20);
  int x = n & 63, y = n >> 6;
  const float* p = src + ((long long)b * CCH + c) * 1024;
  float sx = 0.5f * (float)x - 0.25f;
  float sy = 0.5f * (float)y - 0.25f;
  float fx0 = floorf(sx), fy0 = floorf(sy);
  float wx = sx - fx0, wy = sy - fy0;
  int x0 = (int)fx0, y0 = (int)fy0;
  int x1 = x0 + 1, y1 = y0 + 1;
  x0 = x0 < 0 ? 0 : (x0 > 31 ? 31 : x0);
  x1 = x1 < 0 ? 0 : (x1 > 31 ? 31 : x1);
  y0 = y0 < 0 ? 0 : (y0 > 31 ? 31 : y0);
  y1 = y1 < 0 ? 0 : (y1 > 31 ? 31 : y1);
  float p00 = p[y0 * 32 + x0], p01 = p[y0 * 32 + x1];
  float p10 = p[y1 * 32 + x0], p11 = p[y1 * 32 + x1];
  float v = (1.f - wy) * ((1.f - wx) * p00 + wx * p01)
          +        wy  * ((1.f - wx) * p10 + wx * p11);
  dst[idx] = (half_t)v;
}

// ---------------------------------------------------------------- TDM staging

#if USE_TDM
// Issue one 2-D TDM tile load: 32 halfs x `rows`, global row stride
// `stride` elements, into LDS at byte offset lds_off with 16B pad per 64B row
// (pad_interval code 3 = 64B, pad_amount code 3 = 4 DWORDs) -> 80B LDS stride.
__device__ __forceinline__ void tdm_load_tile(unsigned lds_off,
                                              const half_t* gptr,
                                              int rows, long long stride) {
  unsigned long long ga = (unsigned long long)(uintptr_t)gptr;
  u32x4 g0;
  g0[0] = 1u;                                        // count=1 valid user D#
  g0[1] = lds_off;                                   // LDS byte address
  g0[2] = (unsigned)(ga & 0xFFFFFFFFull);
  g0[3] = (unsigned)((ga >> 32) & 0x01FFFFFFull) | (2u << 30);  // type=2
  const unsigned td = 0x7FFFFFFFu;                   // huge tensor dims: no OOB
  i32x8 g1;
  g1[0] = (1 << 16) | (1 << 20) | (3 << 22) | (3 << 25);  // 2B, pad 64B->+16B
  g1[1] = (int)((td & 0xFFFFu) << 16);               // abar=0 | tdim0 lo16
  g1[2] = (int)((td >> 16) | ((td & 0xFFFFu) << 16));// tdim0 hi16 | tdim1 lo16
  g1[3] = (int)((td >> 16) | (32u << 16));           // tdim1 hi16 | tile_dim0=32
  g1[4] = rows & 0xFFFF;                             // tile_dim1 | tile_dim2=0
  unsigned long long st = (unsigned long long)stride;
  g1[5] = (int)(st & 0xFFFFFFFFull);                 // dim0_stride lo32
  g1[6] = (int)((st >> 32) & 0xFFFFull);             // dim0_stride hi16
  g1[7] = 0;
  i32x4 z4 = {0, 0, 0, 0};
#if TDM_6ARG
  i32x8 z8 = {0, 0, 0, 0, 0, 0, 0, 0};
  __builtin_amdgcn_tensor_load_to_lds(g0, g1, z4, z4, z8, 0);
#else
  __builtin_amdgcn_tensor_load_to_lds(g0, g1, z4, z4, 0);
#endif
}
#endif

// ---------------------------------------------------------------- WMMA GEMM
// C[M,N] = A[M,K] * B[K,N] with B given transposed: Bt[n][k] (k contiguous).
// f16 in / f32 accumulate. Block tile 128x128x32, 8 waves, wave = 2x4 WMMA
// tiles of v_wmma_f32_16x16x32_f16. LDS tiles are linear [row][k] (80B pitch)
// so fragments load as two ds_load_b128 per matrix per tile.

enum { EPI_PHI = 0, EPI_BIAS = 1, EPI_F16 = 2, EPI_DIVNORM = 3, EPI_RESID = 4 };

template <int EPI, bool TRANS>
__global__ void __launch_bounds__(256)
k_gemm_wmma(const half_t* __restrict__ A, long long a_batch, int lda,
            const half_t* __restrict__ Bt, long long b_batch, int ldb,
            int M, int N, int K,
            const float* __restrict__ bias,
            const float* __restrict__ normv,
            const float* __restrict__ resid, long long resid_batch,
            half_t* __restrict__ outh, long long outh_batch,
            float* __restrict__ outf, long long outf_batch) {
  __shared__ half_t As[128 * LDS_STRIDE];
  __shared__ half_t Bs[128 * LDS_STRIDE];

  const int bz = blockIdx.z;
  const half_t* Ab = A  + (long long)bz * a_batch;
  const half_t* Bb = Bt + (long long)bz * b_batch;
  const int m0 = blockIdx.y * 128;
  const int n0 = blockIdx.x * 128;
  const int t    = threadIdx.x;
  const int lane = t & 31;
  const int wave = t >> 5;
  const int wm = (wave & 3) * 32;      // 4 waves tile M
  const int wn = (wave >> 2) * 64;     // 2 waves tile N
  const int lh = lane & 15;
  const int kb = (lane >> 4) * 8;      // K-phase of upper lane half

  v8f acc[2][4];
#pragma unroll
  for (int i = 0; i < 2; i++)
#pragma unroll
    for (int j = 0; j < 4; j++)
#pragma unroll
      for (int v = 0; v < 8; v++) acc[i][j][v] = 0.0f;

  for (int k0 = 0; k0 < K; k0 += 32) {
#if USE_TDM
    if (wave == 0) {
      tdm_load_tile((unsigned)(uintptr_t)&As[0],
                    Ab + (long long)m0 * lda + k0, 128, lda);
      tdm_load_tile((unsigned)(uintptr_t)&Bs[0],
                    Bb + (long long)n0 * ldb + k0, 128, ldb);
      __builtin_amdgcn_s_wait_tensorcnt(0);
    }
    __syncthreads();
#else
    // Manual staging: 512 chunks of 8 halfs per tile; all b128 traffic.
#pragma unroll
    for (int i = 0; i < 2; i++) {
      int id = t + i * 256;            // 0..511
      int row = id >> 2, seg = id & 3;
      *(v8h*)(As + row * LDS_STRIDE + seg * 8) =
          *(const v8h*)(Ab + (long long)(m0 + row) * lda + k0 + seg * 8);
      *(v8h*)(Bs + row * LDS_STRIDE + seg * 8) =
          *(const v8h*)(Bb + (long long)(n0 + row) * ldb + k0 + seg * 8);
    }
    __syncthreads();
#endif

    // Fragments: element i -> K = (i&7) + (i>=8 ? 16 : 0) + (lane>=16 ? 8 : 0)
    v16h bf[4];
#pragma unroll
    for (int tn = 0; tn < 4; tn++) {
      const half_t* pb = Bs + (wn + tn * 16 + lh) * LDS_STRIDE + kb;
      v8h b0 = *(const v8h*)pb;
      v8h b1 = *(const v8h*)(pb + 16);
#pragma unroll
      for (int i = 0; i < 8; i++) { bf[tn][i] = b0[i]; bf[tn][i + 8] = b1[i]; }
    }
#pragma unroll
    for (int tm = 0; tm < 2; tm++) {
      const half_t* pa = As + (wm + tm * 16 + lh) * LDS_STRIDE + kb;
      v8h a0 = *(const v8h*)pa;
      v8h a1 = *(const v8h*)(pa + 16);
      v16h af;
#pragma unroll
      for (int i = 0; i < 8; i++) { af[i] = a0[i]; af[i + 8] = a1[i]; }
#pragma unroll
      for (int tn = 0; tn < 4; tn++) {
        acc[tm][tn] = __builtin_amdgcn_wmma_f32_16x16x32_f16(
            false, af, false, bf[tn], (short)0, acc[tm][tn], false, false);
      }
    }
    __syncthreads();
  }

  // Epilogue. C layout: lanes 0-15 rows v, lanes 16-31 rows v+8; col = lane&15.
  // TRANS path: per lane the 8 rows are contiguous -> single v8h store.
#pragma unroll
  for (int tm = 0; tm < 2; tm++) {
#pragma unroll
    for (int tn = 0; tn < 4; tn++) {
      const int mb = m0 + wm + tm * 16 + ((lane >> 4) << 3);
      const int n  = n0 + wn + tn * 16 + lh;
      float nv = 1.0f;
      if (EPI == EPI_DIVNORM) nv = normv[(long long)bz * N + n];
      if (TRANS) {
        v8h hv;
#pragma unroll
        for (int v = 0; v < 8; v++) {
          float x = acc[tm][tn][v];
          if (EPI == EPI_PHI) {
            x += bias[mb + v];
            x = (x > 0.0f) ? (x + 1.0f) : __expf(x);
          } else if (EPI == EPI_BIAS) {
            x += bias[mb + v];
          } else if (EPI == EPI_DIVNORM) {
            x /= nv;
          }
          hv[v] = (half_t)x;
        }
        *(v8h*)(outh + (long long)bz * outh_batch + (long long)n * M + mb) = hv;
      } else {
#pragma unroll
        for (int v = 0; v < 8; v++) {
          int m = mb + v;
          long long off = (long long)m * N + n;
          float x = acc[tm][tn][v];
          if (EPI == EPI_PHI) {
            x += bias[m];
            x = (x > 0.0f) ? (x + 1.0f) : __expf(x);
            outh[(long long)bz * outh_batch + off] = (half_t)x;
          } else if (EPI == EPI_BIAS) {
            x += bias[m];
            outh[(long long)bz * outh_batch + off] = (half_t)x;
          } else if (EPI == EPI_F16) {
            outh[(long long)bz * outh_batch + off] = (half_t)x;
          } else if (EPI == EPI_DIVNORM) {
            outh[(long long)bz * outh_batch + off] = (half_t)(x / nv);
          } else {  // EPI_RESID -> f32 + residual
            x += bias[m];
            x += resid[(long long)bz * resid_batch + off];
            outf[(long long)bz * outf_batch + off] = x;
          }
        }
      }
    }
  }
}

// ---------------------------------------------------------------- reductions

// K_sum[b,c] = sum_n K[b,c,n]  (K stored [c][n]; deterministic tree reduce)
__global__ void __launch_bounds__(256)
k_ksum(const half_t* __restrict__ Kh, float* __restrict__ ksum) {
  __shared__ float sbuf[256];
  long long row = blockIdx.x;  // b*256 + c
  const half_t* p = Kh + row * (long long)NPIX;
  float s = 0.f;
  for (int i = threadIdx.x; i < NPIX; i += 256) s += (float)p[i];
  float tot = block_reduce_sum(s, sbuf);
  if (threadIdx.x == 0) ksum[row] = tot;
}

// norm[b,n] = sum_c Qt[b,n,c] * Ksum[b,c] + eps (per-thread contiguous reads)
__global__ void __launch_bounds__(256)
k_norm(const half_t* __restrict__ Qt, const float* __restrict__ ksum,
       float* __restrict__ nrm) {
  int idx = blockIdx.x * 256 + threadIdx.x;  // b*4096 + n
  int b = idx >> 12;
  const half_t* q = Qt + (long long)idx * CCH;
  const float* ks = ksum + b * CCH;
  float s = 0.f;
#pragma unroll 8
  for (int c = 0; c < CCH; c++) s += (float)q[c] * ks[c];
  nrm[idx] = s + EPS_ATTN;
}

// GroupNorm(1): two-stage deterministic mean/var over C*H*W per batch
__global__ void __launch_bounds__(256)
k_gn_partial(const float* __restrict__ Y, float* __restrict__ part) {
  __shared__ float sbuf[256];
  int b   = blockIdx.x >> 6;
  int blk = blockIdx.x & 63;
  const float* p = Y + (long long)b * CN + (long long)blk * (CN / 64);
  float s = 0.f, sq = 0.f;
  for (int i = threadIdx.x; i < (int)(CN / 64); i += 256) {
    float v = p[i];
    s += v;
    sq += v * v;
  }
  float ts = block_reduce_sum(s, sbuf);
  float tq = block_reduce_sum(sq, sbuf);
  if (threadIdx.x == 0) {
    part[blockIdx.x * 2 + 0] = ts;
    part[blockIdx.x * 2 + 1] = tq;
  }
}

__global__ void __launch_bounds__(64)
k_gn_final(const float* __restrict__ part, float* __restrict__ stats) {
  __shared__ float ss[64];
  __shared__ float sq[64];
  int b = blockIdx.x;
  int t = threadIdx.x;
  ss[t] = part[(b * 64 + t) * 2 + 0];
  sq[t] = part[(b * 64 + t) * 2 + 1];
  __syncthreads();
  for (int s = 32; s > 0; s >>= 1) {
    if (t < s) { ss[t] += ss[t + s]; sq[t] += sq[t + s]; }
    __syncthreads();
  }
  if (t == 0) {
    const float inv = 1.0f / (float)CN;
    float mean = ss[0] * inv;
    float var  = sq[0] * inv - mean * mean;
    stats[b * 2 + 0] = mean;
    stats[b * 2 + 1] = rsqrtf(var + EPS_GN);
  }
}

__global__ void __launch_bounds__(256)
k_gn_apply(const float* __restrict__ Y, const float* __restrict__ stats,
           const float* __restrict__ w, const float* __restrict__ bia,
           float* __restrict__ out) {
  long long idx = (long long)blockIdx.x * 256 + threadIdx.x;
  int b = (int)(idx >> 20);
  int c = (int)((idx >> 12) & 255);
  float mean = stats[b * 2 + 0], rstd = stats[b * 2 + 1];
  out[idx] = (Y[idx] - mean) * rstd * w[c] + bia[c];
}

// ---------------------------------------------------------------- launcher

extern "C" void kernel_launch(void* const* d_in, const int* in_sizes, int n_in,
                              void* d_out, int out_size, void* d_ws, size_t ws_size,
                              hipStream_t stream) {
  (void)in_sizes; (void)n_in; (void)out_size; (void)ws_size;
  const float* f_geo = (const float*)d_in[0];
  const float* f_sem = (const float*)d_in[1];
  const float* Wq = (const float*)d_in[2];
  const float* bq = (const float*)d_in[3];
  const float* Wk = (const float*)d_in[4];
  const float* bk = (const float*)d_in[5];
  const float* Wv = (const float*)d_in[6];
  const float* bv = (const float*)d_in[7];
  const float* Wo = (const float*)d_in[8];
  const float* bo = (const float*)d_in[9];
  const float* gnw = (const float*)d_in[10];
  const float* gnb = (const float*)d_in[11];
  float* out = (float*)d_out;

  // Workspace layout (256B aligned; ODt overlays XgeoT, Y overlays K|V)
  char* ws = (char*)d_ws;
  size_t off = 0;
  auto alloc = [&](size_t bytes) -> char* {
    char* p = ws + off;
    off += (bytes + 255) & ~(size_t)255;
    return p;
  };
  half_t* XgeoT = (half_t*)alloc(BCN * 2);   // [B][N][C]; reused as ODt
  half_t* XsemT = (half_t*)alloc(BCN * 2);   // [B][N][C]
  half_t* Qt    = (half_t*)alloc(BCN * 2);   // [B][N][C]
  half_t* Kh    = (half_t*)alloc(BCN * 2);   // [B][C][N]; Y overlays Kh..Vh
  half_t* Vh    = (half_t*)alloc(BCN * 2);   // [B][C][N]
  half_t* KVt   = (half_t*)alloc((size_t)BATCH * CCH * CCH * 2);  // [B][D][C]
  half_t* Wqh   = (half_t*)alloc(CCH * CCH * 2);
  half_t* Wkh   = (half_t*)alloc(CCH * CCH * 2);
  half_t* Wvh   = (half_t*)alloc(CCH * CCH * 2);
  half_t* Woh   = (half_t*)alloc(CCH * CCH * 2);
  float*  ksum  = (float*)alloc(BATCH * CCH * 4);
  float*  nrm   = (float*)alloc((size_t)BATCH * NPIX * 4);
  float*  part  = (float*)alloc(BATCH * 64 * 2 * 4);
  float*  stat  = (float*)alloc(BATCH * 2 * 4);
  half_t* ODt   = XgeoT;           // [B][N][D], overlays XgeoT (dead by then)
  float*  Yf    = (float*)Kh;      // [B][C][N] f32, overlays Kh+Vh

  const int WN = CCH * CCH;  // 65536
  // 1) conversions: weights stay [Cout][Cin] (k contiguous), activations go T
  k_f32_to_f16<<<WN / 256, 256, 0, stream>>>(Wq, Wqh, WN);
  k_f32_to_f16<<<WN / 256, 256, 0, stream>>>(Wk, Wkh, WN);
  k_f32_to_f16<<<WN / 256, 256, 0, stream>>>(Wv, Wvh, WN);
  k_f32_to_f16<<<WN / 256, 256, 0, stream>>>(Wo, Woh, WN);
  k_transpose_to_f16<<<(int)(BCN / 256), 256, 0, stream>>>(f_geo, XgeoT);
  k_upsample_t<<<(int)(BCN / 256), 256, 0, stream>>>(f_sem, XsemT);

  dim3 gBig(NPIX / 128, CCH / 128, BATCH);   // (32, 2, 8)
  dim3 gKV(CCH / 128, CCH / 128, BATCH);     // (2, 2, 8)
  const long long CC = (long long)CCH * CCH;

  // 2) Q^T = phi(Wq Xgeo + bq)^T ; K = phi(Wk Xsem + bk) ; V = Wv Xsem + bv
  k_gemm_wmma<EPI_PHI, true><<<gBig, 256, 0, stream>>>(
      Wqh, 0, CCH, XgeoT, CN, CCH, CCH, NPIX, CCH,
      bq, nullptr, nullptr, 0, Qt, CN, nullptr, 0);
  k_gemm_wmma<EPI_PHI, false><<<gBig, 256, 0, stream>>>(
      Wkh, 0, CCH, XsemT, CN, CCH, CCH, NPIX, CCH,
      bk, nullptr, nullptr, 0, Kh, CN, nullptr, 0);
  k_gemm_wmma<EPI_BIAS, false><<<gBig, 256, 0, stream>>>(
      Wvh, 0, CCH, XsemT, CN, CCH, CCH, NPIX, CCH,
      bv, nullptr, nullptr, 0, Vh, CN, nullptr, 0);

  // 3) K_sum over pixels
  k_ksum<<<BATCH * CCH, 256, 0, stream>>>(Kh, ksum);

  // 4) KV^T[d][c] : A=K [c][pix], Bt=V [d][pix], K-dim = 4096, store transposed
  k_gemm_wmma<EPI_F16, true><<<gKV, 256, 0, stream>>>(
      Kh, CN, NPIX, Vh, CN, NPIX, CCH, CCH, NPIX,
      nullptr, nullptr, nullptr, 0, KVt, CC, nullptr, 0);

  // 5) norm = Q^T K_sum + eps ; OD^T = ((KV^T Q) / norm)^T
  k_norm<<<(BATCH * NPIX) / 256, 256, 0, stream>>>(Qt, ksum, nrm);
  k_gemm_wmma<EPI_DIVNORM, true><<<gBig, 256, 0, stream>>>(
      KVt, CC, CCH, Qt, CN, CCH, CCH, NPIX, CCH,
      nullptr, nrm, nullptr, 0, ODt, CN, nullptr, 0);

  // 6) y = f_geo + Wo OD + bo  (f32, [C][N])
  k_gemm_wmma<EPI_RESID, false><<<gBig, 256, 0, stream>>>(
      Woh, 0, CCH, ODt, CN, CCH, CCH, NPIX, CCH,
      bo, nullptr, f_geo, CN, nullptr, 0, Yf, CN);

  // 7) GroupNorm(1) per batch
  k_gn_partial<<<BATCH * 64, 256, 0, stream>>>(Yf, part);
  k_gn_final<<<BATCH, 64, 0, stream>>>(part, stat);
  k_gn_apply<<<(int)(BCN / 256), 256, 0, stream>>>(Yf, stat, gnw, gnb, out);
}
